// CausalSelfAttention_41927470743731
// MI455X (gfx1250) — matrix operand order repaired and checked by reading, not compiled, not run on hardware
//
#include <hip/hip_runtime.h>

// ---------------------------------------------------------------------------
// CDNA5 (gfx1250, wave32) causal self-attention forward.
// All GEMMs use v_wmma_f32_16x16x32_f16 (f16 inputs, f32 accumulate).
// ---------------------------------------------------------------------------

typedef __attribute__((ext_vector_type(16))) _Float16 v16h;
typedef __attribute__((ext_vector_type(8)))  _Float16 v8h;
typedef __attribute__((ext_vector_type(8)))  float    v8f;

constexpr int Bn  = 8;
constexpr int Tn  = 1024;
constexpr int Cn  = 768;
constexpr int Hn  = 12;
constexpr int Dn  = 64;
constexpr int N3C = 3 * Cn;        // 2304

// GEMM tiling: 256 threads = 8 waves, wave grid 2(M) x 4(N),
// wave tile 32x64 (= 2x4 WMMA 16x16 accumulators), block tile 64x256, K-step 32.
constexpr int BM  = 64;
constexpr int BN  = 256;
constexpr int BK  = 32;
constexpr int LDP = 40;            // padded LDS row length in halfs (80B -> bank spread)

__device__ __forceinline__ v16h cat16(v8h lo, v8h hi) {
  return __builtin_shufflevector(lo, hi, 0,1,2,3,4,5,6,7,8,9,10,11,12,13,14,15);
}

__device__ __forceinline__ v8f wmma_f16(v16h a, v16h b, v8f c) {
  // (neg_a, A, neg_b, B, c_mod, C, reuse_a, reuse_b)
  return __builtin_amdgcn_wmma_f32_16x16x32_f16(false, a, false, b, (short)0, c,
                                                false, false);
}

// ---------------------------------------------------------------------------
// Precision staging
// ---------------------------------------------------------------------------
__global__ void f32_to_f16_kernel(const float* __restrict__ src,
                                  _Float16* __restrict__ dst, int n) {
  int i = blockIdx.x * blockDim.x + threadIdx.x;
  if (i < n) dst[i] = (_Float16)src[i];
}

// src: f32 [K, N] row-major  ->  dst: f16 [N, K] row-major
__global__ void transpose_to_f16_kernel(const float* __restrict__ src,
                                        _Float16* __restrict__ dst,
                                        int K, int N) {
  int i = blockIdx.x * blockDim.x + threadIdx.x;
  if (i < N * K) {
    int n = i / K, k = i - n * K;
    dst[i] = (_Float16)src[(size_t)k * N + n];
  }
}

// ---------------------------------------------------------------------------
// Shared GEMM core:  C[64x256 block] += A[M,768] * Bt[N,768]^T
// A row-major [M, 768] f16, Bt row-major [N, 768] f16 (i.e. B transposed).
// ---------------------------------------------------------------------------
__device__ __forceinline__ void gemm_core(const _Float16* __restrict__ A,
                                          const _Float16* __restrict__ Bt,
                                          int mblock, int nblock,
                                          _Float16* As, _Float16* Bs,
                                          v8f acc[2][4]) {
  const int tid  = threadIdx.x;
  const int lane = tid & 31, hi = lane >> 4, ln = lane & 15;
  const int wave = tid >> 5;
  const int wm = wave & 1, wn = wave >> 1;
  const int arow = tid >> 2, acol = (tid & 3) << 3;

  const _Float16* ag = A  + (size_t)(mblock + arow) * Cn + acol;
  const _Float16* bg = Bt + (size_t)(nblock + tid ) * Cn;

  for (int mi = 0; mi < 2; ++mi)
    for (int ni = 0; ni < 4; ++ni)
      for (int e = 0; e < 8; ++e) acc[mi][ni][e] = 0.0f;

  for (int k0 = 0; k0 < Cn; k0 += BK) {
    // Stage A tile (64x32) and Bt tile (256x32) in LDS, vectorized b128.
    *(v8h*)&As[arow * LDP + acol] = *(const v8h*)(ag + k0);
    const _Float16* bgk = bg + k0;
    *(v8h*)&Bs[tid * LDP +  0] = *(const v8h*)(bgk +  0);
    *(v8h*)&Bs[tid * LDP +  8] = *(const v8h*)(bgk +  8);
    *(v8h*)&Bs[tid * LDP + 16] = *(const v8h*)(bgk + 16);
    *(v8h*)&Bs[tid * LDP + 24] = *(const v8h*)(bgk + 24);
    __syncthreads();

    // A fragment (16x32): lane row = ln, elems 0-7 = K 8*hi.., elems 8-15 = K 16+8*hi..
    v16h af[2];
    for (int mi = 0; mi < 2; ++mi) {
      const _Float16* p = &As[(wm * 32 + mi * 16 + ln) * LDP + 8 * hi];
      af[mi] = cat16(*(const v8h*)p, *(const v8h*)(p + 16));
    }
    // B fragment (32x16): lane col = ln, elems 0-7 = K 8*hi.., elems 8-15 = K 16+8*hi..
    v16h bf[4];
    for (int ni = 0; ni < 4; ++ni) {
      const _Float16* p = &Bs[(wn * 64 + ni * 16 + ln) * LDP + 8 * hi];
      bf[ni] = cat16(*(const v8h*)p, *(const v8h*)(p + 16));
    }
    for (int mi = 0; mi < 2; ++mi)
      for (int ni = 0; ni < 4; ++ni)
        acc[mi][ni] = wmma_f16(af[mi], bf[ni], acc[mi][ni]);
    __syncthreads();
  }
}

// ---------------------------------------------------------------------------
// QKV GEMM: [8192,768] x [768,2304] + bias, scatter into
//   Q,K : f16 [B,H,T,D]      V : f16 [B,H,D,T] (transposed for P@V B-frags)
// ---------------------------------------------------------------------------
__global__ __launch_bounds__(256) void qkv_gemm_kernel(
    const _Float16* __restrict__ xf, const _Float16* __restrict__ wqkvT,
    const float* __restrict__ b_qkv, _Float16* __restrict__ qf,
    _Float16* __restrict__ kf, _Float16* __restrict__ vtf) {
  __shared__ _Float16 As[BM * LDP];
  __shared__ _Float16 Bs[BN * LDP];
  v8f acc[2][4];
  const int mblock = blockIdx.y * BM, nblock = blockIdx.x * BN;
  gemm_core(xf, wqkvT, mblock, nblock, As, Bs, acc);

  const int tid = threadIdx.x, lane = tid & 31, hi = lane >> 4, ln = lane & 15;
  const int wave = tid >> 5, wm = wave & 1, wn = wave >> 1;
  for (int mi = 0; mi < 2; ++mi) {
    const int mbase = mblock + wm * 32 + mi * 16 + 8 * hi;
    for (int ni = 0; ni < 4; ++ni) {
      const int n = nblock + wn * 64 + ni * 16 + ln;
      const float bias = b_qkv[n];
      const int which = n / Cn, c = n % Cn, h = c >> 6, d = c & 63;
      for (int r = 0; r < 8; ++r) {
        const int m = mbase + r;                 // global row in [0, B*T)
        const int b = m >> 10, t = m & 1023;
        const _Float16 hv = (_Float16)(acc[mi][ni][r] + bias);
        const size_t bh = (size_t)(b * Hn + h);
        if (which == 0)      qf [(bh * Tn + t) * Dn + d] = hv;
        else if (which == 1) kf [(bh * Tn + t) * Dn + d] = hv;
        else                 vtf[(bh * Dn + d) * Tn + t] = hv;
      }
    }
  }
}

// ---------------------------------------------------------------------------
// Flash attention: grid (T/64, B*H), 128 threads = 4 independent waves.
// Each wave owns 16 query rows; key chunks of 32 (2 S-tiles + 4 PV WMMAs).
// ---------------------------------------------------------------------------
__global__ __launch_bounds__(128) void attn_kernel(
    const _Float16* __restrict__ qf, const _Float16* __restrict__ kf,
    const _Float16* __restrict__ vtf, _Float16* __restrict__ yf) {
  __shared__ _Float16 Ps[4][16 * LDP];          // per-wave P staging (f32->f16 relayout)

  const int tid = threadIdx.x, wave = tid >> 5, lane = tid & 31;
  const int hi = lane >> 4, ln = lane & 15;
  const int bh  = blockIdx.y;                   // b*H + h
  const int qt0 = blockIdx.x * 64 + wave * 16;  // first query row of this wave
  const size_t base = (size_t)bh * Tn * Dn;
  const _Float16* Q  = qf  + base;
  const _Float16* Kp = kf  + base;
  const _Float16* Vt = vtf + base;              // [D, T]

  // Q A-fragments (held in registers): qa0 = d 0..31, qa1 = d 32..63
  v16h qa0, qa1;
  {
    const _Float16* p = &Q[(size_t)(qt0 + ln) * Dn + 8 * hi];
    qa0 = cat16(*(const v8h*)p,        *(const v8h*)(p + 16));
    qa1 = cat16(*(const v8h*)(p + 32), *(const v8h*)(p + 48));
  }

  v8f o[4];
  for (int n = 0; n < 4; ++n)
    for (int e = 0; e < 8; ++e) o[n][e] = 0.0f;
  float mrow[8], lrow[8];
  for (int r = 0; r < 8; ++r) { mrow[r] = -__builtin_inff(); lrow[r] = 0.0f; }
  const float scale = 0.125f;                   // 1/sqrt(64)

  for (int j0 = 0; j0 <= qt0 + 15; j0 += 32) {
    // ---- S = Q K^T for two 16-key tiles --------------------------------
    v8f s[2];
    for (int t = 0; t < 2; ++t) {
      const _Float16* kp = &Kp[(size_t)(j0 + t * 16 + ln) * Dn];
      v16h kb0 = cat16(*(const v8h*)(kp + 8 * hi),      *(const v8h*)(kp + 8 * hi + 16));
      v16h kb1 = cat16(*(const v8h*)(kp + 32 + 8 * hi), *(const v8h*)(kp + 32 + 8 * hi + 16));
      v8f sv;
      for (int e = 0; e < 8; ++e) sv[e] = 0.0f;
      sv = wmma_f16(qa0, kb0, sv);
      sv = wmma_f16(qa1, kb1, sv);
      s[t] = sv;
    }
    // ---- scale + causal mask (diagonal chunks only) --------------------
    const bool needmask = (j0 + 31 > qt0);
    for (int t = 0; t < 2; ++t)
      for (int r = 0; r < 8; ++r) {
        float v = s[t][r] * scale;
        if (needmask) {
          const int key = j0 + t * 16 + ln;
          const int q   = qt0 + r + 8 * hi;
          if (key > q) v = -__builtin_inff();
        }
        s[t][r] = v;
      }
    // ---- online softmax (row stats across 16-lane half-wave) -----------
    float rmax[8];
    for (int r = 0; r < 8; ++r) rmax[r] = fmaxf(s[0][r], s[1][r]);
    for (int off = 8; off > 0; off >>= 1)
      for (int r = 0; r < 8; ++r)
        rmax[r] = fmaxf(rmax[r], __shfl_xor(rmax[r], off, 16));

    float sf[8], rsum[8];
    for (int r = 0; r < 8; ++r) {
      const float nm = fmaxf(mrow[r], rmax[r]);
      sf[r]   = __expf(mrow[r] - nm);
      mrow[r] = nm;
      const float p0 = __expf(s[0][r] - nm);
      const float p1 = __expf(s[1][r] - nm);
      s[0][r] = p0; s[1][r] = p1;
      rsum[r] = p0 + p1;
    }
    for (int off = 8; off > 0; off >>= 1)
      for (int r = 0; r < 8; ++r)
        rsum[r] += __shfl_xor(rsum[r], off, 16);
    for (int r = 0; r < 8; ++r) lrow[r] = lrow[r] * sf[r] + rsum[r];
    for (int n = 0; n < 4; ++n)
      for (int r = 0; r < 8; ++r) o[n][r] *= sf[r];

    // ---- relayout P (f32 C-layout) -> f16 A-fragment via per-wave LDS --
    _Float16* pl = &Ps[wave][0];
    for (int t = 0; t < 2; ++t)
      for (int r = 0; r < 8; ++r)
        pl[(r + 8 * hi) * LDP + t * 16 + ln] = (_Float16)s[t][r];
    // Waves have divergent trip counts -> no block barrier; DS ops are
    // in-order per wave, just drain the DS counter before re-reading.
    asm volatile("s_wait_dscnt 0x0" ::: "memory");
    const _Float16* pp = &pl[ln * LDP + 8 * hi];
    const v16h pa = cat16(*(const v8h*)pp, *(const v8h*)(pp + 16));

    // ---- O += P @ V  (V transposed: B-frag rows contiguous in t) -------
    for (int n = 0; n < 4; ++n) {
      const _Float16* vp = &Vt[(size_t)(n * 16 + ln) * Tn + j0 + 8 * hi];
      const v16h vb = cat16(*(const v8h*)vp, *(const v8h*)(vp + 16));
      o[n] = wmma_f16(pa, vb, o[n]);
    }
  }

  // ---- normalize and store y as f16 [B,T,C] ----------------------------
  const int b = bh / Hn, h = bh % Hn;
  for (int r = 0; r < 8; ++r) {
    const float inv = 1.0f / lrow[r];
    const int t = qt0 + r + 8 * hi;
    for (int n = 0; n < 4; ++n) {
      const int d = n * 16 + ln;
      yf[((size_t)(b * Tn + t)) * Cn + h * Dn + d] = (_Float16)(o[n][r] * inv);
    }
  }
}

// ---------------------------------------------------------------------------
// Output projection: [8192,768] x [768,768] + bias -> f32 out
// ---------------------------------------------------------------------------
__global__ __launch_bounds__(256) void proj_gemm_kernel(
    const _Float16* __restrict__ yf, const _Float16* __restrict__ wprojT,
    const float* __restrict__ b_proj, float* __restrict__ out) {
  __shared__ _Float16 As[BM * LDP];
  __shared__ _Float16 Bs[BN * LDP];
  v8f acc[2][4];
  const int mblock = blockIdx.y * BM, nblock = blockIdx.x * BN;
  gemm_core(yf, wprojT, mblock, nblock, As, Bs, acc);

  const int tid = threadIdx.x, lane = tid & 31, hi = lane >> 4, ln = lane & 15;
  const int wave = tid >> 5, wm = wave & 1, wn = wave >> 1;
  for (int mi = 0; mi < 2; ++mi) {
    const int mbase = mblock + wm * 32 + mi * 16 + 8 * hi;
    for (int ni = 0; ni < 4; ++ni) {
      const int n = nblock + wn * 64 + ni * 16 + ln;
      const float bias = b_proj[n];
      for (int r = 0; r < 8; ++r)
        out[(size_t)(mbase + r) * Cn + n] = acc[mi][ni][r] + bias;
    }
  }
}

// ---------------------------------------------------------------------------
// Host launcher
// ---------------------------------------------------------------------------
extern "C" void kernel_launch(void* const* d_in, const int* in_sizes, int n_in,
                              void* d_out, int out_size, void* d_ws, size_t ws_size,
                              hipStream_t stream) {
  (void)in_sizes; (void)n_in; (void)out_size; (void)ws_size;
  const float* x      = (const float*)d_in[0];
  const float* w_qkv  = (const float*)d_in[1];
  const float* b_qkv  = (const float*)d_in[2];
  const float* w_proj = (const float*)d_in[3];
  const float* b_proj = (const float*)d_in[4];
  float* out = (float*)d_out;

  char* ws = (char*)d_ws;
  size_t off = 0;
  auto take = [&](size_t bytes) -> char* {
    char* p = ws + off;
    off = (off + bytes + 255) & ~(size_t)255;
    return p;
  };
  const size_t BTC = (size_t)Bn * Tn * Cn;            // 6,291,456
  _Float16* xf     = (_Float16*)take(BTC * 2);
  _Float16* wqkvT  = (_Float16*)take((size_t)N3C * Cn * 2);
  _Float16* wprojT = (_Float16*)take((size_t)Cn * Cn * 2);
  _Float16* qf     = (_Float16*)take(BTC * 2);
  _Float16* kf     = (_Float16*)take(BTC * 2);
  _Float16* vtf    = (_Float16*)take(BTC * 2);
  _Float16* yf     = (_Float16*)take(BTC * 2);

  // Stage precisions / layouts.
  f32_to_f16_kernel<<<(int)((BTC + 255) / 256), 256, 0, stream>>>(x, xf, (int)BTC);
  transpose_to_f16_kernel<<<(N3C * Cn + 255) / 256, 256, 0, stream>>>(w_qkv, wqkvT, Cn, N3C);
  transpose_to_f16_kernel<<<(Cn * Cn + 255) / 256, 256, 0, stream>>>(w_proj, wprojT, Cn, Cn);

  // QKV projection.
  qkv_gemm_kernel<<<dim3(N3C / BN, (Bn * Tn) / BM), 256, 0, stream>>>(
      xf, wqkvT, b_qkv, qf, kf, vtf);

  // Flash attention.
  attn_kernel<<<dim3(Tn / 64, Bn * Hn), 128, 0, stream>>>(qf, kf, vtf, yf);

  // Output projection.
  proj_gemm_kernel<<<dim3(Cn / BN, (Bn * Tn) / BM), 256, 0, stream>>>(
      yf, wprojT, b_proj, out);
}
